// SequenceAttention_67061619360437
// MI455X (gfx1250) — matrix operand order repaired and checked by reading, not compiled, not run on hardware
//
#include <hip/hip_runtime.h>
#include <hip/hip_bf16.h>
#include <stdint.h>

// Shapes (fixed by the reference): B=8, Q=64, K=1024, D=256, A=256
#define BB 8
#define QQ 64
#define KK 1024
#define DD 256
#define AA 256
#define MROWS (BB*KK + BB*QQ)   // 8704 projection rows (x rows then query rows)

typedef __attribute__((ext_vector_type(16))) __bf16 v16bf;
typedef __attribute__((ext_vector_type(8)))  float  v8f;

// Split an f32 into bf16 hi + bf16 lo so that 3 bf16 WMMAs reproduce ~f32 GEMM accuracy.
__device__ inline void split16(const float* f, v16bf& hi, v16bf& lo) {
#pragma unroll
  for (int i = 0; i < 16; ++i) {
    __bf16 h = (__bf16)f[i];
    hi[i] = h;
    lo[i] = (__bf16)(f[i] - (float)h);
  }
}

__device__ inline void unpack4(float4 v, float* f) {
  f[0] = v.x; f[1] = v.y; f[2] = v.z; f[3] = v.w;
}

// Native V_TANH_F32 if the toolchain exposes it (gfx1250 TRANS op); else
// synthesize from v_exp_f32 + v_rcp_f32: tanh(x) = sign(x)*(1 - 2/(exp(2|x|)+1))
__device__ inline float fast_tanh(float v) {
#if __has_builtin(__builtin_amdgcn_tanhf)
  return __builtin_amdgcn_tanhf(v);
#elif __has_builtin(__builtin_amdgcn_tanh_f32)
  return __builtin_amdgcn_tanh_f32(v);
#else
  float ax = __builtin_fabsf(v);
  float e  = __builtin_amdgcn_exp2f(ax * 2.8853900817779268f);  // 2*log2(e)
  float r  = 1.0f - 2.0f * __builtin_amdgcn_rcpf(e + 1.0f);
  return __builtin_copysignf(r, v);
#endif
}

// ---------------------------------------------------------------------------
// Kernel 1: kxq[r, a] = (r < 8192 ? x[r] . W[a, 0:256] : query[r-8192] . W[a, 256:512] + bias[a])
// One 16x16 output tile per wave, K=256 contraction in 8 steps of 32, 3 WMMAs/step.
// ---------------------------------------------------------------------------
__global__ void __launch_bounds__(256)
kernel_proj(const float* __restrict__ x, const float* __restrict__ query,
            const float* __restrict__ W, const float* __restrict__ bias,
            float* __restrict__ kxq) {
  const int lane  = threadIdx.x & 31;
  const int wave  = threadIdx.x >> 5;
  const int rowTile = blockIdx.x;              // 0..543
  const int aTile   = blockIdx.y * 8 + wave;   // 0..15
  const int rowBase = rowTile * 16;
  const int aBase   = aTile * 16;
  const int lmod = lane & 15;
  const int lhi  = lane >> 4;                  // lane group (0/1)

  const bool isQ = (rowBase >= BB*KK);
  const int  r   = rowBase + lmod;             // this lane's A-matrix row
  const float* aRow = isQ ? (query + (size_t)(r - BB*KK) * DD)
                          : (x + (size_t)r * DD);
  const float* bRow = W + (size_t)(aBase + lmod) * (2*DD) + (isQ ? DD : 0);

  v8f acc = {};
  const int aoff = lhi ? 16 : 0;  // A-matrix per-lane-group K offset (contiguous, matches B)

#pragma unroll
  for (int db = 0; db < 8; ++db) {
    const int d0 = db * 32;
    float af[16], bfv[16];
    // A k order chosen here (contiguous): lane group 0 -> K 0..15, group 1 -> K 16..31
    const float4* pa0 = (const float4*)(aRow + d0 + aoff);
    const float4* pa1 = (const float4*)(aRow + d0 + aoff + 8);
    unpack4(pa0[0], af + 0);  unpack4(pa0[1], af + 4);
    unpack4(pa1[0], af + 8);  unpack4(pa1[1], af + 12);
    // B k order chosen here (contiguous, same as A): lanes 0-15 hold K=0..15 (col n=lane), lanes 16-31 K=16..31
    const float4* pb = (const float4*)(bRow + d0 + lhi * 16);
    unpack4(pb[0], bfv + 0);  unpack4(pb[1], bfv + 4);
    unpack4(pb[2], bfv + 8);  unpack4(pb[3], bfv + 12);

    v16bf ah, al, bh, bl;
    split16(af, ah, al);
    split16(bfv, bh, bl);
    acc = __builtin_amdgcn_wmma_f32_16x16x32_bf16(false, ah, false, bh, (short)0, acc, false, false);
    acc = __builtin_amdgcn_wmma_f32_16x16x32_bf16(false, ah, false, bl, (short)0, acc, false, false);
    acc = __builtin_amdgcn_wmma_f32_16x16x32_bf16(false, al, false, bh, (short)0, acc, false, false);
  }

  if (isQ) {  // fold bias into the qx rows: tanh(kx + qx + b)
    float bv = bias[aBase + lmod];  // N = lane%16
#pragma unroll
    for (int i = 0; i < 8; ++i) acc[i] += bv;
  }

  // C/D layout: VGPR i -> row rowBase + i + 8*(lane/16), col aBase + lane%16
  float* dst = kxq + (size_t)(rowBase + lhi * 8) * AA + aBase + lmod;
#pragma unroll
  for (int i = 0; i < 8; ++i) dst[(size_t)i * AA] = acc[i];
}

// ---------------------------------------------------------------------------
// Kernel 2: scores[b,q,k] = sum_a w2[a]*tanh(kx[b,k,a] + qxb[b,q,a]); mask -> -inf
// Tile: one WG = (b, 16 q's, 16 k's). kx/qx tiles staged in LDS via async-LDS loads.
// Thread (q,k) does the full 256-wide reduction from LDS (rows padded to kill conflicts).
// ---------------------------------------------------------------------------
__global__ void __launch_bounds__(256)
kernel_scores(const float* __restrict__ kxq, const float* __restrict__ w2,
              const unsigned char* __restrict__ mask, float* __restrict__ scores) {
  __shared__ float kxs[16][260];
  __shared__ float qxs[16][260];
  __shared__ float w2s[256];
  const int kt = blockIdx.x;   // 0..63
  const int qt = blockIdx.y;   // 0..3
  const int b  = blockIdx.z;   // 0..7
  const int t  = threadIdx.x;
  const int lr = t >> 4;            // load row 0..15
  const int cb = (t & 15) * 16;     // load col base

  const float* gk = kxq + (size_t)(b * KK + kt * 16 + lr) * AA + cb;
  const float* gq = kxq + (size_t)(BB * KK + b * QQ + qt * 16 + lr) * AA + cb;
#pragma unroll
  for (int i = 0; i < 4; ++i) {
    unsigned ldk = (unsigned)(uintptr_t)&kxs[lr][cb + i * 4];
    asm volatile("global_load_async_to_lds_b128 %0, %1, off"
                 :: "v"(ldk), "v"(gk + i * 4) : "memory");
    unsigned ldq = (unsigned)(uintptr_t)&qxs[lr][cb + i * 4];
    asm volatile("global_load_async_to_lds_b128 %0, %1, off"
                 :: "v"(ldq), "v"(gq + i * 4) : "memory");
  }
  w2s[t] = w2[t];
  asm volatile("s_wait_asynccnt 0x0" ::: "memory");
  __syncthreads();

  const int k_l = t & 15;
  const int q_l = t >> 4;
  const float4* kr4 = (const float4*)&kxs[k_l][0];
  const float4* qr4 = (const float4*)&qxs[q_l][0];
  const float4* w24 = (const float4*)&w2s[0];
  float s = 0.f;
#pragma unroll 8
  for (int i = 0; i < AA / 4; ++i) {
    float4 kv = kr4[i], qv = qr4[i], wv = w24[i];
    s += wv.x * fast_tanh(kv.x + qv.x);
    s += wv.y * fast_tanh(kv.y + qv.y);
    s += wv.z * fast_tanh(kv.z + qv.z);
    s += wv.w * fast_tanh(kv.w + qv.w);
  }

  const int kg = kt * 16 + k_l;
  const int qg = qt * 16 + q_l;
  if (mask[b * KK + kg]) s = -__builtin_inff();
  scores[(size_t)(b * QQ + qg) * KK + kg] = s;
}

// ---------------------------------------------------------------------------
// Kernel 3: in-place masked softmax over K=1024 per (b,q) row.
// ---------------------------------------------------------------------------
__global__ void __launch_bounds__(256)
kernel_softmax(float* __restrict__ scores) {
  __shared__ float red[256];
  const int t = threadIdx.x;
  float* sr = scores + (size_t)blockIdx.x * KK;
  float v0 = sr[t], v1 = sr[t + 256], v2 = sr[t + 512], v3 = sr[t + 768];
  float m = fmaxf(fmaxf(v0, v1), fmaxf(v2, v3));
  red[t] = m;
  __syncthreads();
  for (int s2 = 128; s2 > 0; s2 >>= 1) {
    if (t < s2) red[t] = fmaxf(red[t], red[t + s2]);
    __syncthreads();
  }
  m = red[0];
  __syncthreads();
  const float L2E = 1.4426950408889634f;
  float e0 = __builtin_amdgcn_exp2f((v0 - m) * L2E);   // -inf -> 0
  float e1 = __builtin_amdgcn_exp2f((v1 - m) * L2E);
  float e2 = __builtin_amdgcn_exp2f((v2 - m) * L2E);
  float e3 = __builtin_amdgcn_exp2f((v3 - m) * L2E);
  red[t] = e0 + e1 + e2 + e3;
  __syncthreads();
  for (int s2 = 128; s2 > 0; s2 >>= 1) {
    if (t < s2) red[t] += red[t + s2];
    __syncthreads();
  }
  float inv = __builtin_amdgcn_rcpf(red[0]);
  sr[t] = e0 * inv; sr[t + 256] = e1 * inv; sr[t + 512] = e2 * inv; sr[t + 768] = e3 * inv;
}

// ---------------------------------------------------------------------------
// Kernel 4: out[b] = P[b](64x1024) @ x[b](1024x256), split-precision bf16 WMMA.
// ---------------------------------------------------------------------------
__global__ void __launch_bounds__(256)
kernel_av(const float* __restrict__ probs, const float* __restrict__ x,
          float* __restrict__ out) {
  const int lane  = threadIdx.x & 31;
  const int wave  = threadIdx.x >> 5;
  const int nTile = blockIdx.x * 8 + wave;  // 0..15
  const int mTile = blockIdx.y;             // 0..3
  const int b     = blockIdx.z;             // 0..7
  const int lmod = lane & 15;
  const int lhi  = lane >> 4;

  const float* pRow = probs + (size_t)(b * QQ + mTile * 16 + lmod) * KK;
  const int n = nTile * 16 + lmod;
  const float* xcol = x + (size_t)b * KK * DD + n;

  v8f acc = {};
  const int aoff = lhi ? 16 : 0;
#pragma unroll 2
  for (int kb = 0; kb < KK / 32; ++kb) {
    const int k0 = kb * 32;
    float af[16], bfv[16];
    const float4* pa0 = (const float4*)(pRow + k0 + aoff);
    const float4* pa1 = (const float4*)(pRow + k0 + aoff + 8);
    unpack4(pa0[0], af + 0);  unpack4(pa0[1], af + 4);
    unpack4(pa1[0], af + 8);  unpack4(pa1[1], af + 12);
    const float* xb = xcol + (size_t)(k0 + lhi * 16) * DD;
#pragma unroll
    for (int j = 0; j < 16; ++j) bfv[j] = xb[(size_t)j * DD];

    v16bf ah, al, bh, bl;
    split16(af, ah, al);
    split16(bfv, bh, bl);
    acc = __builtin_amdgcn_wmma_f32_16x16x32_bf16(false, ah, false, bh, (short)0, acc, false, false);
    acc = __builtin_amdgcn_wmma_f32_16x16x32_bf16(false, ah, false, bl, (short)0, acc, false, false);
    acc = __builtin_amdgcn_wmma_f32_16x16x32_bf16(false, al, false, bh, (short)0, acc, false, false);
  }

  float* dst = out + (size_t)(b * QQ + mTile * 16 + lhi * 8) * DD + n;
#pragma unroll
  for (int i = 0; i < 8; ++i) dst[(size_t)i * DD] = acc[i];
}

// ---------------------------------------------------------------------------
extern "C" void kernel_launch(void* const* d_in, const int* in_sizes, int n_in,
                              void* d_out, int out_size, void* d_ws, size_t ws_size,
                              hipStream_t stream) {
  const float*         x     = (const float*)d_in[0];
  const unsigned char* mask  = (const unsigned char*)d_in[1];  // jnp bool_ -> 1 byte
  const float*         query = (const float*)d_in[2];
  const float*         W     = (const float*)d_in[3];
  const float*         bias  = (const float*)d_in[4];
  const float*         w2    = (const float*)d_in[5];
  float*               out   = (float*)d_out;

  // workspace: kxq (8704*256 f32 = 8.5 MB) then scores/probs (8*64*1024 f32 = 2 MB)
  float* kxq    = (float*)d_ws;
  float* scores = kxq + (size_t)MROWS * AA;

  kernel_proj   <<<dim3(MROWS / 16, 2, 1), 256, 0, stream>>>(x, query, W, bias, kxq);
  kernel_scores <<<dim3(KK / 16, QQ / 16, BB), 256, 0, stream>>>(kxq, w2, mask, scores);
  kernel_softmax<<<dim3(BB * QQ, 1, 1), 256, 0, stream>>>(scores);
  kernel_av     <<<dim3(2, QQ / 16, BB), 256, 0, stream>>>(scores, x, out);
}
